// MultiTaskVulnerabilityGNN_9122510537366
// MI455X (gfx1250) — matrix operand order repaired and checked by reading, not compiled, not run on hardware
//
#include <hip/hip_runtime.h>
#include <math.h>

// ---------------------------------------------------------------------------
// MultiTaskVulnerabilityGNN for MI455X (gfx1250, wave32, WMMA).
// GEMMs: v_wmma_f32_16x16x32_f16, double-buffered LDS tiles filled with
// CDNA5 async global->LDS copies (ASYNCcnt), weights pre-transposed so both
// tiles stage as b128 with no in-kernel transpose.
// Edge softmax / aggregation use native float global atomics.
// ---------------------------------------------------------------------------

typedef __attribute__((ext_vector_type(16))) _Float16 v16h;
typedef __attribute__((ext_vector_type(8)))  _Float16 v8h;
typedef __attribute__((ext_vector_type(8)))  float    v8f;

#define TILE_M 128
#define TILE_N 64
#define TILE_K 32
#define AS_STRIDE 40   // 32 + 8 halves pad (80B/row, keeps 16B alignment)

// ---- CDNA5 async global->LDS copy (per-lane, EXEC-masked, ASYNCcnt) ------
__device__ __forceinline__ void async_b128(void* lds_ptr, const void* gptr) {
    // Generic shared pointers are {aperture_hi, lds_offset}; low 32 bits are
    // the LDS byte offset the instruction's VDST operand expects.
    unsigned loff = (unsigned)(unsigned long long)lds_ptr;
    asm volatile("global_load_async_to_lds_b128 %0, %1, off"
                 :: "v"(loff), "v"(gptr) : "memory");
}
__device__ __forceinline__ void wait_async() {
    asm volatile("s_wait_asynccnt 0x0" ::: "memory");
}

// ------------------------------ WMMA GEMM ----------------------------------
// C[M,Nc] = A[M,K] (f16 row major) @ B (given pre-TRANSPOSED: Bt[Nc,K] f16),
// f32 out. Block: 256 threads = 8 waves (4x2); each wave computes 32x32
// (2x2 fragments). Requires K % 32 == 0 and Nc % 64 == 0.
__global__ __launch_bounds__(256) void wmma_gemm_f16(
    const _Float16* __restrict__ A, const _Float16* __restrict__ Bt,
    float* __restrict__ C, const float* __restrict__ bias,
    int M, int K, int Nc, int doRelu)
{
    __shared__ __align__(16) _Float16 As[2][TILE_M * AS_STRIDE];
    __shared__ __align__(16) _Float16 Bs[2][TILE_N * TILE_K]; // [col][k]

    const int tid  = threadIdx.x;
    const int wid  = tid >> 5;
    const int lane = tid & 31;
    const int wr   = wid & 3;     // wave row band (0..3)
    const int wc   = wid >> 2;    // wave col band (0..1)
    const int kg   = lane >> 4;   // lane group (ISA 16-bit fragment layout)
    const int l15  = lane & 15;
    const int m0   = blockIdx.x * TILE_M;
    const int n0   = blockIdx.y * TILE_N;

    v8f acc[2][2] = {};

    // stage tiles for k-chunk kc into buffer `buf` (async, no wait here)
    auto stage = [&](int buf, int kc) {
        #pragma unroll
        for (int i = 0; i < 2; ++i) {
            int l   = tid + i * 256;
            int row = l >> 2;
            int ch  = l & 3;
            _Float16* dst = &As[buf][row * AS_STRIDE + ch * 8];
            int gr = m0 + row;
            if (gr < M) {
                async_b128(dst, A + (size_t)gr * K + kc + ch * 8);
            } else {
                v8h z = {};
                *(v8h*)dst = z;                 // zero-fill OOB rows (DScnt)
            }
        }
        {
            int col = tid >> 2;
            int ch  = tid & 3;
            _Float16* dst = &Bs[buf][col * TILE_K + ch * 8];
            async_b128(dst, Bt + (size_t)(n0 + col) * K + kc + ch * 8);
        }
    };

    const int nk = K / TILE_K;
    stage(0, 0);
    wait_async();
    __syncthreads();

    for (int t = 0; t < nk; ++t) {
        int cur = t & 1;
        if (t + 1 < nk) stage(cur ^ 1, (t + 1) * TILE_K);  // overlap with WMMA

        // ---- build fragments per ISA 16-bit A(16x32)/B(32x16) layouts ----
        v16h af[2], bf[2];
        #pragma unroll
        for (int fm = 0; fm < 2; ++fm) {
            const _Float16* Ar = &As[cur][(wr * 32 + fm * 16 + l15) * AS_STRIDE];
            v8h lo = *(const v8h*)(Ar + kg * 8);        // K = kg*8 .. +7
            v8h hi = *(const v8h*)(Ar + 16 + kg * 8);   // K = 16+kg*8 .. +7
            af[fm] = __builtin_shufflevector(lo, hi, 0,1,2,3,4,5,6,7,
                                                     8,9,10,11,12,13,14,15);
        }
        #pragma unroll
        for (int fn = 0; fn < 2; ++fn) {
            const _Float16* Bc = &Bs[cur][(wc * 32 + fn * 16 + l15) * TILE_K
                                          + kg * 8];
            v8h lo = *(const v8h*)(Bc);                 // K = kg*8 .. +7
            v8h hi = *(const v8h*)(Bc + 16);            // K = 16+kg*8 .. +7
            bf[fn] = __builtin_shufflevector(lo, hi, 0,1,2,3,4,5,6,7,
                                                     8,9,10,11,12,13,14,15);
        }
        #pragma unroll
        for (int fm = 0; fm < 2; ++fm)
            #pragma unroll
            for (int fn = 0; fn < 2; ++fn)
                acc[fm][fn] = __builtin_amdgcn_wmma_f32_16x16x32_f16(
                    false, af[fm], false, bf[fn], (short)0, acc[fm][fn],
                    false, false);

        wait_async();        // next-iter tiles landed (this wave's issues)
        __syncthreads();     // all waves' tiles landed / reads of cur done
    }

    // ---- epilogue: D layout = lane (l15) -> col, kg*8+vgpr -> row ----
    #pragma unroll
    for (int fm = 0; fm < 2; ++fm) {
        #pragma unroll
        for (int fn = 0; fn < 2; ++fn) {
            int col = n0 + wc * 32 + fn * 16 + l15;
            float bv = bias ? bias[col] : 0.f;
            #pragma unroll
            for (int vg = 0; vg < 8; ++vg) {
                int row = m0 + wr * 32 + fm * 16 + kg * 8 + vg;
                if (row < M) {
                    float v = acc[fm][fn][vg] + bv;
                    if (doRelu) v = v > 0.f ? v : 0.f;
                    C[(size_t)row * Nc + col] = v;
                }
            }
        }
    }
}

// ------------------------------ utilities ----------------------------------
__global__ void fill_kernel(float* p, float v, size_t count) {
    size_t i = (size_t)blockIdx.x * blockDim.x + threadIdx.x;
    if (i < count) p[i] = v;
}

__global__ void cvt_f16_kernel(const float* __restrict__ src,
                               _Float16* __restrict__ dst, size_t count) {
    size_t i = (size_t)blockIdx.x * blockDim.x + threadIdx.x;
    if (i < count) dst[i] = (_Float16)src[i];
}

// weights: src f32 [K,Nc] row-major -> dst f16 [Nc,K] (transposed)
__global__ void cvt_f16_t_kernel(const float* __restrict__ src,
                                 _Float16* __restrict__ dst, int K, int Nc) {
    int i = blockIdx.x * blockDim.x + threadIdx.x;
    if (i >= K * Nc) return;
    int k = i / Nc, n = i - k * Nc;
    dst[(size_t)n * K + k] = (_Float16)src[i];
}

// --------------------------- attention logits ------------------------------
// als[n,h] = sum_f h[n,h,f] * a_src[h,f]; one wave32 per node.
__global__ __launch_bounds__(256) void attn_kernel(
    const float* __restrict__ h, const float* __restrict__ a_src,
    const float* __restrict__ a_dst, float* __restrict__ als,
    float* __restrict__ ald, int n, int heads, int hid)
{
    int wid = threadIdx.x >> 5, lane = threadIdx.x & 31;
    int node = blockIdx.x * 8 + wid;
    if (node >= n) return;
    float ss[4] = {0.f, 0.f, 0.f, 0.f};
    float sd[4] = {0.f, 0.f, 0.f, 0.f};
    int Dl = heads * hid;
    for (int d = lane; d < Dl; d += 32) {
        float v = h[(size_t)node * Dl + d];
        int hh = d / hid;
        ss[hh] += v * a_src[d];
        sd[hh] += v * a_dst[d];
    }
    for (int hh = 0; hh < heads; ++hh) {
        float a = ss[hh], b = sd[hh];
        for (int off = 16; off; off >>= 1) {
            a += __shfl_xor(a, off, 32);
            b += __shfl_xor(b, off, 32);
        }
        if (lane == 0) {
            als[(size_t)node * heads + hh] = a;
            ald[(size_t)node * heads + hh] = b;
        }
    }
}

// -------------------------- edge softmax pieces ----------------------------
__device__ __forceinline__ void atomicMaxF(float* addr, float v) {
    if (v >= 0.f) atomicMax((int*)addr, __float_as_int(v));
    else          atomicMin((unsigned int*)addr, __float_as_uint(v));
}

__device__ __forceinline__ void edge_ends(const int* ei, int E, int j,
                                          int& s, int& d) {
    if (j < E) { s = ei[j]; d = ei[E + j]; }
    else       { s = d = j - E; }            // self loops appended
}

__global__ void edge_max_kernel(const float* __restrict__ als,
                                const float* __restrict__ ald,
                                const int* __restrict__ ei,
                                float* __restrict__ m, int E, int n, int heads)
{
    int j = blockIdx.x * blockDim.x + threadIdx.x;
    if (j >= E + n) return;
    int s, d; edge_ends(ei, E, j, s, d);
    for (int hh = 0; hh < heads; ++hh) {
        float e = als[(size_t)s * heads + hh] + ald[(size_t)d * heads + hh];
        e = e > 0.f ? e : 0.2f * e;          // leaky_relu, slope 0.2
        atomicMaxF(&m[(size_t)d * heads + hh], e);
    }
}

__global__ void edge_exp_kernel(const float* __restrict__ als,
                                const float* __restrict__ ald,
                                const int* __restrict__ ei,
                                const float* __restrict__ m,
                                float* __restrict__ exbuf,
                                float* __restrict__ denom,
                                int E, int n, int heads)
{
    int j = blockIdx.x * blockDim.x + threadIdx.x;
    if (j >= E + n) return;
    int s, d; edge_ends(ei, E, j, s, d);
    for (int hh = 0; hh < heads; ++hh) {
        float e = als[(size_t)s * heads + hh] + ald[(size_t)d * heads + hh];
        e = e > 0.f ? e : 0.2f * e;
        float ex = expf(e - m[(size_t)d * heads + hh]);
        exbuf[(size_t)j * heads + hh] = ex;
        atomicAdd(&denom[(size_t)d * heads + hh], ex);
    }
}

// out[dst] += h[src] * alpha  (one block per edge, threads over features)
__global__ __launch_bounds__(256) void edge_agg_kernel(
    const float* __restrict__ h, const float* __restrict__ exbuf,
    const float* __restrict__ denom, const int* __restrict__ ei,
    float* __restrict__ out, int E, int n, int D, int heads)
{
    int j = blockIdx.x;
    int s, d; edge_ends(ei, E, j, s, d);
    int hid = D / heads;
    for (int f = threadIdx.x; f < D; f += blockDim.x) {
        int hh = f / hid;
        float alpha = exbuf[(size_t)j * heads + hh]
                    / (denom[(size_t)d * heads + hh] + 1e-16f);
        atomicAdd(&out[(size_t)d * D + f], h[(size_t)s * D + f] * alpha);
    }
}

// ------------------------------ batch norm ---------------------------------
__global__ __launch_bounds__(256) void colstats_kernel(
    const float* __restrict__ x, float* __restrict__ sums,
    float* __restrict__ sumsq, int n, int D)
{
    int r0 = blockIdx.x * 128;
    for (int c = threadIdx.x; c < D; c += 256) {
        float s = 0.f, q = 0.f;
        for (int r = 0; r < 128; ++r) {
            int row = r0 + r;
            if (row < n) {
                float v = x[(size_t)row * D + c];
                s += v; q += v * v;
            }
        }
        atomicAdd(&sums[c], s);
        atomicAdd(&sumsq[c], q);
    }
}

// y = (x - mu) * rsqrt(var + eps) * g + be ; optional relu; also emits f16.
__global__ void bn_apply_kernel(float* __restrict__ x, _Float16* __restrict__ xh,
                                const float* __restrict__ sums,
                                const float* __restrict__ sumsq,
                                const float* __restrict__ g,
                                const float* __restrict__ be,
                                int n, int D, int doRelu)
{
    size_t i = (size_t)blockIdx.x * blockDim.x + threadIdx.x;
    size_t total = (size_t)n * D;
    if (i >= total) return;
    int col = (int)(i % D);
    float invn = 1.f / (float)n;
    float mu  = sums[col] * invn;
    float var = sumsq[col] * invn - mu * mu;
    float v = (x[i] - mu) * rsqrtf(var + 1e-5f) * g[col] + be[col];
    if (doRelu) v = v > 0.f ? v : 0.f;
    x[i]  = v;
    xh[i] = (_Float16)v;
}

// ------------------------- MLP head final reduce ---------------------------
// logits[n, c] = dot(H1[n, :dh], w2) + b2 ; one wave32 per node.
__global__ __launch_bounds__(256) void head_reduce_kernel(
    const float* __restrict__ H1, const float* __restrict__ w2,
    const float* __restrict__ b2, float* __restrict__ out,
    int n, int dh, int c, int nc)
{
    int wid = threadIdx.x >> 5, lane = threadIdx.x & 31;
    int node = blockIdx.x * 8 + wid;
    if (node >= n) return;
    float s = 0.f;
    for (int d = lane; d < dh; d += 32)
        s += H1[(size_t)node * dh + d] * w2[d];
    for (int off = 16; off; off >>= 1) s += __shfl_xor(s, off, 32);
    if (lane == 0) out[(size_t)node * nc + c] = s + b2[0];
}

// ------------------------------- launcher ----------------------------------
extern "C" void kernel_launch(void* const* d_in, const int* in_sizes, int n_in,
                              void* d_out, int out_size, void* d_ws,
                              size_t ws_size, hipStream_t stream)
{
    (void)in_sizes; (void)n_in; (void)out_size; (void)ws_size;
    const int N = 20000, E = 320000, ET = E + N, DIN = 128, D = 512;
    const int NC = 5, DH = 256;

    const float* x   = (const float*)d_in[0];
    const int*   ei  = (const int*)d_in[1];     // int32 (JAX x64 disabled)
    const float* W0  = (const float*)d_in[2];
    const float* as0 = (const float*)d_in[3];
    const float* ad0 = (const float*)d_in[4];
    // d_in[5]  = b0 : cancels through batch norm (and is zero)
    const float* g0  = (const float*)d_in[6];
    const float* be0 = (const float*)d_in[7];
    const float* W1  = (const float*)d_in[8];
    const float* as1 = (const float*)d_in[9];
    const float* ad1 = (const float*)d_in[10];
    const float* g1  = (const float*)d_in[12];
    const float* be1 = (const float*)d_in[13];
    const float* W2  = (const float*)d_in[14];
    const float* as2 = (const float*)d_in[15];
    const float* ad2 = (const float*)d_in[16];
    const float* g2  = (const float*)d_in[18];
    const float* be2 = (const float*)d_in[19];
    const float* cweW1 = (const float*)d_in[20];
    const float* cweB1 = (const float*)d_in[21];
    const float* cweW2 = (const float*)d_in[22];
    const float* cweB2 = (const float*)d_in[23];
    const float* trgW1 = (const float*)d_in[24];
    const float* trgB1 = (const float*)d_in[25];
    const float* trgW2 = (const float*)d_in[26];
    const float* trgB2 = (const float*)d_in[27];
    float* out = (float*)d_out;

    // ---- workspace carve-out ----
    char* base = (char*)d_ws;
    size_t off = 0;
    auto take = [&](size_t bytes) -> char* {
        size_t o = (off + 255) & ~(size_t)255;
        off = o + bytes;
        return base + o;
    };
    _Float16* xh  = (_Float16*)take((size_t)N * D * 2);   // GEMM A operand (f16)
    _Float16* wh  = (_Float16*)take((size_t)D * D * 2);   // B^T operand (f16)
    float* hbuf   = (float*)take((size_t)N * D * 4);      // pre-agg features
    float* hagg   = (float*)take((size_t)N * D * 4);      // aggregated features
    float* als    = (float*)take((size_t)N * 4 * 4);
    float* ald    = (float*)take((size_t)N * 4 * 4);
    float* mbuf   = (float*)take((size_t)N * 4 * 4);
    float* den    = (float*)take((size_t)N * 4 * 4);
    float* exb    = (float*)take((size_t)ET * 4 * 4);
    float* sums   = (float*)take((size_t)D * 4);
    float* sq     = (float*)take((size_t)D * 4);
    float* Htmp   = (float*)take((size_t)N * DH * 4);

    auto gl = [](size_t count) { return dim3((unsigned)((count + 255) / 256)); };

    // initial f16 activations
    cvt_f16_kernel<<<gl((size_t)N * DIN), 256, 0, stream>>>(x, xh, (size_t)N * DIN);

    auto run_conv = [&](int K, const float* W, const float* asrc,
                        const float* adst, const float* g, const float* be,
                        int heads, int doRelu) {
        int hid = D / heads;
        cvt_f16_t_kernel<<<gl((size_t)K * D), 256, 0, stream>>>(W, wh, K, D);
        dim3 gg((N + TILE_M - 1) / TILE_M, D / TILE_N);
        wmma_gemm_f16<<<gg, 256, 0, stream>>>(xh, wh, hbuf, nullptr, N, K, D, 0);
        attn_kernel<<<(N + 7) / 8, 256, 0, stream>>>(hbuf, asrc, adst, als, ald,
                                                     N, heads, hid);
        fill_kernel<<<gl((size_t)N * heads), 256, 0, stream>>>(
            mbuf, -3.4028235e38f, (size_t)N * heads);
        fill_kernel<<<gl((size_t)N * heads), 256, 0, stream>>>(
            den, 0.f, (size_t)N * heads);
        edge_max_kernel<<<(ET + 255) / 256, 256, 0, stream>>>(als, ald, ei, mbuf,
                                                              E, N, heads);
        edge_exp_kernel<<<(ET + 255) / 256, 256, 0, stream>>>(als, ald, ei, mbuf,
                                                              exb, den, E, N, heads);
        fill_kernel<<<gl((size_t)N * D), 256, 0, stream>>>(hagg, 0.f, (size_t)N * D);
        edge_agg_kernel<<<ET, 256, 0, stream>>>(hbuf, exb, den, ei, hagg,
                                                E, N, D, heads);
        fill_kernel<<<gl(D), 256, 0, stream>>>(sums, 0.f, (size_t)D);
        fill_kernel<<<gl(D), 256, 0, stream>>>(sq, 0.f, (size_t)D);
        colstats_kernel<<<(N + 127) / 128, 256, 0, stream>>>(hagg, sums, sq, N, D);
        bn_apply_kernel<<<gl((size_t)N * D), 256, 0, stream>>>(
            hagg, xh, sums, sq, g, be, N, D, doRelu);
    };

    run_conv(DIN, W0, as0, ad0, g0, be0, 4, 1);   // conv0 + bn + relu
    run_conv(D,   W1, as1, ad1, g1, be1, 4, 1);   // conv1 + bn + relu
    run_conv(D,   W2, as2, ad2, g2, be2, 1, 0);   // conv2 + bn (no relu)

    // ---- 5 CWE + 5 trigger heads: Linear(512,256)+ReLU then dot(256) ----
    for (int task = 0; task < 2; ++task) {
        const float* HW1 = task ? trgW1 : cweW1;
        const float* HB1 = task ? trgB1 : cweB1;
        const float* HW2 = task ? trgW2 : cweW2;
        const float* HB2 = task ? trgB2 : cweB2;
        float* obase = out + (size_t)task * N * NC;
        for (int c = 0; c < NC; ++c) {
            cvt_f16_t_kernel<<<gl((size_t)D * DH), 256, 0, stream>>>(
                HW1 + (size_t)c * D * DH, wh, D, DH);
            dim3 gg((N + TILE_M - 1) / TILE_M, DH / TILE_N);
            wmma_gemm_f16<<<gg, 256, 0, stream>>>(xh, wh, Htmp,
                                                  HB1 + (size_t)c * DH,
                                                  N, D, DH, 1);
            head_reduce_kernel<<<(N + 7) / 8, 256, 0, stream>>>(
                Htmp, HW2 + (size_t)c * DH, HB2 + c, obase, N, DH, c, NC);
        }
    }
}
